// EMAutoencoder_47296179864060
// MI455X (gfx1250) — compile-verified, matrix operand order checked by reading, not run on hardware
//
#include <hip/hip_runtime.h>
#include <hip/hip_bf16.h>

typedef __attribute__((ext_vector_type(16))) __bf16 v16bf;
typedef __attribute__((ext_vector_type(8)))  __bf16 v8bf;
typedef __attribute__((ext_vector_type(8)))  float  v8f;

#define SZ_B 4096
#define SZ_D 2048
#define SZ_L 16384
#define SZ_K 64

// Encode GEMM tiling: block 128x128, 8 waves as 4(M) x 2(N), wave tile 32x64.
#define BM 128
#define BN 128
#define BK 32
#define LDA 40   // padded bf16 stride (80 B/row, keeps 16B-aligned b128 LDS ops)
#define LDB 40

#if defined(__has_builtin)
#if __has_builtin(__builtin_amdgcn_global_load_async_to_lds_b128) && \
    __has_builtin(__builtin_amdgcn_s_wait_asynccnt)
#define USE_ASYNC_LDS 1
#endif
#endif
#ifndef USE_ASYNC_LDS
#define USE_ASYNC_LDS 0
#endif

#if USE_ASYNC_LDS
// Builtin signature (from hipcc diagnostic): (v4i addrspace(1)*, v4i addrspace(3)*, imm, imm)
typedef int i4v __attribute__((vector_size(16)));
typedef __attribute__((address_space(1))) i4v* gbl_i4p;
typedef __attribute__((address_space(3))) i4v* lds_i4p;
__device__ inline void async_cp16(const void* g, void* l) {
  // 16B global -> LDS via the async engine (ASYNCcnt-tracked)
  __builtin_amdgcn_global_load_async_to_lds_b128((gbl_i4p)g, (lds_i4p)l, 0, 0);
}
#endif

// Build one 16x32 bf16 WMMA fragment from LDS, per ISA A/B layout:
// lanes 0-15 -> K {0..7, 16..23}; lanes 16-31 -> K {8..15, 24..31}.
__device__ inline v16bf frag_ld(const __bf16* base, int row, int stride, int lane) {
  const __bf16* p = base + (row + (lane & 15)) * stride + ((lane >> 4) << 3);
  v8bf lo = *(const v8bf*)(p);
  v8bf hi = *(const v8bf*)(p + 16);
  v16bf f;
#pragma unroll
  for (int i = 0; i < 8; ++i) { f[i] = lo[i]; f[i + 8] = hi[i]; }
  return f;
}

// ---------------- pre-pass: x (f32) -> bf16 ----------------
__global__ __launch_bounds__(256)
void cvt_x_bf16(const float* __restrict__ src, __bf16* __restrict__ dst)
{
  size_t i = ((size_t)blockIdx.x * 256 + threadIdx.x) * 8;
  const float4* p = (const float4*)(src + i);
  float4 f0 = p[0], f1 = p[1];
  v8bf h;
  h[0] = (__bf16)f0.x; h[1] = (__bf16)f0.y; h[2] = (__bf16)f0.z; h[3] = (__bf16)f0.w;
  h[4] = (__bf16)f1.x; h[5] = (__bf16)f1.y; h[6] = (__bf16)f1.z; h[7] = (__bf16)f1.w;
  *(v8bf*)(dst + i) = h;
}

// ---- pre-pass: W_enc [D,L] f32 -> Wt [L,D] bf16 (tiled transpose via LDS) ----
__global__ __launch_bounds__(256)
void transpose_w_bf16(const float* __restrict__ W, __bf16* __restrict__ Wt)
{
  __shared__ __align__(16) __bf16 T[64 * 72];   // 64x64 tile, padded stride 72
  const int tid = threadIdx.x;
  const int l0 = blockIdx.x * 64;
  const int d0 = blockIdx.y * 64;
  {
    int r = tid >> 2;              // d-row within tile
    int c = (tid & 3) * 16;        // l-col within tile
    const float4* p = (const float4*)(W + (size_t)(d0 + r) * SZ_L + l0 + c);
    float4 f0 = p[0], f1 = p[1], f2 = p[2], f3 = p[3];
    v8bf h0, h1;
    h0[0]=(__bf16)f0.x; h0[1]=(__bf16)f0.y; h0[2]=(__bf16)f0.z; h0[3]=(__bf16)f0.w;
    h0[4]=(__bf16)f1.x; h0[5]=(__bf16)f1.y; h0[6]=(__bf16)f1.z; h0[7]=(__bf16)f1.w;
    h1[0]=(__bf16)f2.x; h1[1]=(__bf16)f2.y; h1[2]=(__bf16)f2.z; h1[3]=(__bf16)f2.w;
    h1[4]=(__bf16)f3.x; h1[5]=(__bf16)f3.y; h1[6]=(__bf16)f3.z; h1[7]=(__bf16)f3.w;
    *(v8bf*)&T[r * 72 + c]     = h0;
    *(v8bf*)&T[r * 72 + c + 8] = h1;
  }
  __syncthreads();
  {
    int n = tid >> 2;              // l-row within tile
    int c = (tid & 3) * 16;        // d-col within tile
    v8bf h0, h1;
#pragma unroll
    for (int j = 0; j < 8; ++j) {
      h0[j] = T[(c + j) * 72 + n];
      h1[j] = T[(c + 8 + j) * 72 + n];
    }
    *(v8bf*)(Wt + (size_t)(l0 + n) * SZ_D + d0 + c)     = h0;
    *(v8bf*)(Wt + (size_t)(l0 + n) * SZ_D + d0 + c + 8) = h1;
  }
}

// ---------------- encode GEMM: z = x @ W_enc + b_enc (bf16 WMMA) ----------------
__global__ __launch_bounds__(256)
void encode_wmma(const __bf16* __restrict__ xbf, const __bf16* __restrict__ wtbf,
                 const float* __restrict__ benc, float* __restrict__ z)
{
  __shared__ __align__(16) __bf16 sA[2][BM * LDA];
  __shared__ __align__(16) __bf16 sB[2][BN * LDB];

  const int tid    = threadIdx.x;
  const int lane   = tid & 31;
  const int wave   = tid >> 5;
  const int wm     = wave & 3;    // M sub-tile of 32
  const int wn     = wave >> 2;   // N sub-tile of 64
  const int blockM = blockIdx.y * BM;
  const int blockN = blockIdx.x * BN;

  // Staging: 128 rows x 32 bf16 = 512 x 16B chunks per tile; 2 A-chunks + 2 B-chunks
  // per thread. Both A and B(=Wt rows) are contiguous row-chunks now.
  const int r0  = tid >> 2;            // 0..63
  const int r1  = r0 + 64;             // 64..127
  const int off = (tid & 3) * 8;       // bf16 element offset {0,8,16,24}

  const __bf16* gA0 = xbf  + (size_t)(blockM + r0) * SZ_D + off;
  const __bf16* gA1 = xbf  + (size_t)(blockM + r1) * SZ_D + off;
  const __bf16* gB0 = wtbf + (size_t)(blockN + r0) * SZ_D + off;
  const __bf16* gB1 = wtbf + (size_t)(blockN + r1) * SZ_D + off;

  __bf16* lA0 = &sA[0][r0 * LDA + off];
  __bf16* lA1 = &sA[0][r1 * LDA + off];
  __bf16* lB0 = &sB[0][r0 * LDB + off];
  __bf16* lB1 = &sB[0][r1 * LDB + off];
  const size_t bufStrideA = (size_t)BM * LDA;   // elements between buffers
  const size_t bufStrideB = (size_t)BN * LDB;

  v8f acc[2][4];
#pragma unroll
  for (int mi = 0; mi < 2; ++mi)
#pragma unroll
    for (int ni = 0; ni < 4; ++ni) acc[mi][ni] = (v8f){};

#if USE_ASYNC_LDS
  // Double-buffered async pipeline: one s_wait_asynccnt + one barrier per K-step.
  async_cp16(gA0, lA0); async_cp16(gA1, lA1);
  async_cp16(gB0, lB0); async_cp16(gB1, lB1);
  int buf = 0;
  for (int k0 = 0; k0 < SZ_D; k0 += BK) {
    __builtin_amdgcn_s_wait_asynccnt(0);
    __syncthreads();                          // tile[buf] complete across all waves
    if (k0 + BK < SZ_D) {                     // prefetch next tile into buf^1
      int nb = buf ^ 1;
      async_cp16(gA0 + k0 + BK, lA0 + nb * bufStrideA);
      async_cp16(gA1 + k0 + BK, lA1 + nb * bufStrideA);
      async_cp16(gB0 + k0 + BK, lB0 + nb * bufStrideB);
      async_cp16(gB1 + k0 + BK, lB1 + nb * bufStrideB);
    }
#else
  int buf = 0;
  for (int k0 = 0; k0 < SZ_D; k0 += BK) {
    uint4 a0 = *(const uint4*)(gA0 + k0);
    uint4 a1 = *(const uint4*)(gA1 + k0);
    uint4 b0 = *(const uint4*)(gB0 + k0);
    uint4 b1 = *(const uint4*)(gB1 + k0);
    __syncthreads();                          // previous tile reads done
    *(uint4*)lA0 = a0; *(uint4*)lA1 = a1;
    *(uint4*)lB0 = b0; *(uint4*)lB1 = b1;
    __syncthreads();                          // tile complete
#endif

    v16bf a0f = frag_ld(&sA[buf][0], wm * 32,      LDA, lane);
    v16bf a1f = frag_ld(&sA[buf][0], wm * 32 + 16, LDA, lane);
    v16bf b0f = frag_ld(&sB[buf][0], wn * 64,      LDB, lane);
    v16bf b1f = frag_ld(&sB[buf][0], wn * 64 + 16, LDB, lane);
    v16bf b2f = frag_ld(&sB[buf][0], wn * 64 + 32, LDB, lane);
    v16bf b3f = frag_ld(&sB[buf][0], wn * 64 + 48, LDB, lane);

    acc[0][0] = __builtin_amdgcn_wmma_f32_16x16x32_bf16(false, a0f, false, b0f, (short)0, acc[0][0], false, false);
    acc[0][1] = __builtin_amdgcn_wmma_f32_16x16x32_bf16(false, a0f, false, b1f, (short)0, acc[0][1], false, false);
    acc[0][2] = __builtin_amdgcn_wmma_f32_16x16x32_bf16(false, a0f, false, b2f, (short)0, acc[0][2], false, false);
    acc[0][3] = __builtin_amdgcn_wmma_f32_16x16x32_bf16(false, a0f, false, b3f, (short)0, acc[0][3], false, false);
    acc[1][0] = __builtin_amdgcn_wmma_f32_16x16x32_bf16(false, a1f, false, b0f, (short)0, acc[1][0], false, false);
    acc[1][1] = __builtin_amdgcn_wmma_f32_16x16x32_bf16(false, a1f, false, b1f, (short)0, acc[1][1], false, false);
    acc[1][2] = __builtin_amdgcn_wmma_f32_16x16x32_bf16(false, a1f, false, b2f, (short)0, acc[1][2], false, false);
    acc[1][3] = __builtin_amdgcn_wmma_f32_16x16x32_bf16(false, a1f, false, b3f, (short)0, acc[1][3], false, false);

#if USE_ASYNC_LDS
    buf ^= 1;
  }
#else
  }
#endif

  // Epilogue: C/D layout -> lanes 0-15: VGPR r = (M=r, N=lane); lanes 16-31: (M=r+8, N=lane-16)
  const int lcol    = lane & 15;
  const int lrow    = (lane >> 4) * 8;
  const int rowBase = blockM + wm * 32;
  const int colBase = blockN + wn * 64;
#pragma unroll
  for (int ni = 0; ni < 4; ++ni) {
    int col = colBase + ni * 16 + lcol;
    float bias = benc[col];
#pragma unroll
    for (int mi = 0; mi < 2; ++mi) {
      int mb = rowBase + mi * 16 + lrow;
#pragma unroll
      for (int r = 0; r < 8; ++r)
        z[(size_t)(mb + r) * SZ_L + col] = acc[mi][ni][r] + bias;
    }
  }
}

// ---------------- exact abs-top-K mask (deterministic) ----------------
__global__ __launch_bounds__(256)
void topk_mask(float* __restrict__ z, int* __restrict__ idxbuf, float* __restrict__ valbuf)
{
  extern __shared__ unsigned char smem[];
  unsigned int* keys = (unsigned int*)smem;                                  // SZ_L
  int*   red  = (int*)(smem + (size_t)SZ_L * sizeof(unsigned int));          // 256
  int*   sIdx = (int*)((unsigned char*)red + 256 * sizeof(int));             // 64
  float* sVal = (float*)((unsigned char*)sIdx + SZ_K * sizeof(int));         // 64
  int*   ctr  = (int*)((unsigned char*)sVal + SZ_K * sizeof(float));         // 1

  const int tid = threadIdx.x;
  const int row = blockIdx.x;
  float* zrow = z + (size_t)row * SZ_L;

  for (int j = tid; j < SZ_L; j += 256) {
    float v = zrow[j];
    keys[j] = __float_as_uint(fabsf(v));
  }
  __syncthreads();

  // f(t) = count(key > t): smallest t with f(t) < K  ==  K-th largest key
  unsigned int lo = 0u, hi = 0xFFFFFFFFu;
  for (int it = 0; it < 32 && lo < hi; ++it) {
    unsigned int mid = lo + ((hi - lo) >> 1);
    int cnt = 0;
    for (int j = tid; j < SZ_L; j += 256) cnt += (keys[j] > mid) ? 1 : 0;
    red[tid] = cnt; __syncthreads();
    for (int s = 128; s > 0; s >>= 1) { if (tid < s) red[tid] += red[tid + s]; __syncthreads(); }
    int total = red[0]; __syncthreads();
    if (total >= SZ_K) lo = mid + 1; else hi = mid;
  }
  const unsigned int tstar = lo;

  int cnt = 0;
  for (int j = tid; j < SZ_L; j += 256) cnt += (keys[j] > tstar) ? 1 : 0;
  red[tid] = cnt; __syncthreads();
  for (int s = 128; s > 0; s >>= 1) { if (tid < s) red[tid] += red[tid + s]; __syncthreads(); }
  const int G = red[0]; __syncthreads();
  const int quota = SZ_K - G;

  if (tid == 0) ctr[0] = 0;
  if (tid < SZ_K) { sIdx[tid] = 0; sVal[tid] = 0.0f; }
  __syncthreads();

  for (int j = tid; j < SZ_L; j += 256) {
    unsigned int kk = keys[j];
    bool sel = kk > tstar;
    if (!sel && kk == tstar) {
      int rank = 0;                       // rank among equal keys, by index
      for (int i = 0; i < j; ++i) rank += (keys[i] == tstar) ? 1 : 0;
      sel = (rank < quota);
    }
    float v = zrow[j];
    zrow[j] = sel ? v : 0.0f;             // in-place mask -> z_sparse output
    if (sel) {
      int p = atomicAdd(ctr, 1);
      if (p < SZ_K) { sIdx[p] = j; sVal[p] = v; }
    }
  }
  __syncthreads();

  if (tid == 0) {                         // index-sort -> deterministic decode order
    int n = ctr[0]; if (n > SZ_K) n = SZ_K;
    for (int i = 1; i < n; ++i) {
      int ki = sIdx[i]; float kv = sVal[i]; int jj = i - 1;
      while (jj >= 0 && sIdx[jj] > ki) { sIdx[jj + 1] = sIdx[jj]; sVal[jj + 1] = sVal[jj]; --jj; }
      sIdx[jj + 1] = ki; sVal[jj + 1] = kv;
    }
  }
  __syncthreads();
  if (tid < SZ_K) {
    idxbuf[(size_t)row * SZ_K + tid] = sIdx[tid];
    valbuf[(size_t)row * SZ_K + tid] = sVal[tid];
  }
}

// ---------------- sparse decode: W_dec (134 MB) stays L2-resident (192 MB) ----------------
__global__ __launch_bounds__(256)
void decode_sparse(const int* __restrict__ idxbuf, const float* __restrict__ valbuf,
                   const float* __restrict__ Wdec, const float* __restrict__ bdec,
                   float* __restrict__ recon)
{
  __shared__ int   sIdx[SZ_K];
  __shared__ float sVal[SZ_K];
  const int tid = threadIdx.x;
  const int row = blockIdx.x;
  if (tid < SZ_K) {
    sIdx[tid] = idxbuf[(size_t)row * SZ_K + tid];
    sVal[tid] = valbuf[(size_t)row * SZ_K + tid];
  }
  __syncthreads();

  float acc[8];
#pragma unroll
  for (int i = 0; i < 8; ++i) acc[i] = bdec[tid + i * 256];

  for (int t = 0; t < SZ_K; ++t) {
    const float* wrow = Wdec + (size_t)sIdx[t] * SZ_D;
    if (t + 1 < SZ_K) __builtin_prefetch(Wdec + (size_t)sIdx[t + 1] * SZ_D + tid, 0, 0);
    float v = sVal[t];
#pragma unroll
    for (int i = 0; i < 8; ++i) acc[i] = fmaf(v, wrow[tid + i * 256], acc[i]);
  }

  float* out = recon + (size_t)row * SZ_D;
#pragma unroll
  for (int i = 0; i < 8; ++i) out[tid + i * 256] = acc[i];
}

extern "C" void kernel_launch(void* const* d_in, const int* in_sizes, int n_in,
                              void* d_out, int out_size, void* d_ws, size_t ws_size,
                              hipStream_t stream) {
  const float* x    = (const float*)d_in[0];
  const float* Wenc = (const float*)d_in[1];
  const float* benc = (const float*)d_in[2];
  const float* Wdec = (const float*)d_in[3];
  const float* bdec = (const float*)d_in[4];
  (void)in_sizes; (void)n_in; (void)out_size; (void)ws_size;

  float* recon = (float*)d_out;                              // [B, D]
  float* zout  = (float*)d_out + (size_t)SZ_B * SZ_D;        // [B, L] (masked in place)

  unsigned char* ws = (unsigned char*)d_ws;
  int*    idxbuf = (int*)ws;                                                  // 1 MB
  float*  valbuf = (float*)(ws + (size_t)SZ_B * SZ_K * sizeof(int));          // 1 MB
  __bf16* xbf    = (__bf16*)(ws + (size_t)SZ_B * SZ_K * 8);                   // 16 MB
  __bf16* wtbf   = (__bf16*)(ws + (size_t)SZ_B * SZ_K * 8
                                + (size_t)SZ_B * SZ_D * sizeof(__bf16));      // 64 MB

  // One-time precision/layout pre-pass (removes all conversion from the GEMM loop)
  cvt_x_bf16<<<(SZ_B * SZ_D) / (8 * 256), 256, 0, stream>>>(x, xbf);
  transpose_w_bf16<<<dim3(SZ_L / 64, SZ_D / 64), 256, 0, stream>>>(Wenc, wtbf);

  dim3 gEnc(SZ_L / BN, SZ_B / BM);
  encode_wmma<<<gEnc, 256, 0, stream>>>(xbf, wtbf, benc, zout);

  size_t smem = (size_t)SZ_L * 4 + 256 * 4 + SZ_K * 4 + SZ_K * 4 + 16;
  topk_mask<<<SZ_B, 256, smem, stream>>>(zout, idxbuf, valbuf);

  decode_sparse<<<SZ_B, 256, 0, stream>>>(idxbuf, valbuf, Wdec, bdec, recon);
}
